// AFNONet_33105607918127
// MI455X (gfx1250) — compile-verified, matrix operand order checked by reading, not run on hardware
//
#include <hip/hip_runtime.h>
#include <hip/hip_bf16.h>

// ---------------------------------------------------------------------------
// AFNONet forward for MI455X (gfx1250, wave32, WMMA + TDM).
// Dense math -> v_wmma_f32_16x16x32_f16 (f32 accumulate).
// Weights pre-converted to f16 [N][K] once per layer; B tiles DMA'd to LDS by
// the Tensor Data Mover (with hardware pad to the 80B bank-padded row pitch).
// ---------------------------------------------------------------------------

typedef __attribute__((ext_vector_type(16))) _Float16 v16h;
typedef __attribute__((ext_vector_type(8)))  _Float16 v8h;
typedef __attribute__((ext_vector_type(4)))  _Float16 h4;
typedef __attribute__((ext_vector_type(8)))  float    v8f;
typedef __attribute__((ext_vector_type(4)))  unsigned int u32x4;
typedef __attribute__((ext_vector_type(8)))  int          i32x8;
typedef __attribute__((ext_vector_type(4)))  int          i32x4;

union HF { v16h v; v8h h[2]; };

#define GEMM_BM 128
#define GEMM_BN 128
#define GEMM_BK 32
#define GEMM_LDT 40   // BK + 8 halves of padding (80B row pitch, 16B aligned)

// TDM available? (device pass only; fall back to vector loads otherwise)
#if defined(__HIP_DEVICE_COMPILE__) && __has_builtin(__builtin_amdgcn_tensor_load_to_lds) && __has_builtin(__builtin_amdgcn_s_wait_tensorcnt)
#define USE_TDM 1
#else
#define USE_TDM 0
#endif

enum { ACT_NONE = 0, ACT_RELU = 1, ACT_GELU = 2, ACT_SIG = 3, ACT_SSHR = 4 };

template <int ACT>
__device__ __forceinline__ float act_fn(float v) {
  if (ACT == ACT_RELU) return v > 0.f ? v : 0.f;
  if (ACT == ACT_GELU) return 0.5f * v * (1.f + erff(v * 0.70710678118f));
  if (ACT == ACT_SIG)  return 1.f / (1.f + __expf(-v));
  if (ACT == ACT_SSHR) return v > 0.01f ? v - 0.01f : (v < -0.01f ? v + 0.01f : 0.f);
  return v;
}

#if USE_TDM
// Issue a TDM load of one B tile: 128 lines x 64B (32 f16), line stride K*2 B,
// into LDS at lds_off with hardware padding 16B per 64B line -> 80B pitch.
// D# layout per CDNA5 ISA ch.8 (group0 128b, group1 256b; groups 2/3 zero).
__device__ __forceinline__ void tdm_load_b_tile(const _Float16* gptr,
                                                unsigned lds_off, int K) {
  const unsigned long long ga = (unsigned long long)(size_t)gptr;
  u32x4 g0 = {
      1u,                                    // count=1, user descriptor
      lds_off,                               // lds_addr (bytes)
      (unsigned)(ga & 0xffffffffu),          // global_addr[31:0]
      (unsigned)(ga >> 32) | (2u << 30)      // global_addr[56:32] | type=2
  };
  i32x8 g1 = {
      (int)0x06D20000u,   // data_size=4B, pad_enable, pad_interval=16dw, pad_amount=4dw
      (int)(16u << 16),   // tensor_dim0 = 16 units (4B) = 64B line
      (int)(128u << 16),  // tensor_dim0 hi | tensor_dim1 = 128 lines
      (int)(16u << 16),   // tensor_dim1 hi | tile_dim0 = 16 units
      128,                // tile_dim1 = 128, tile_dim2 = 0
      K >> 1,             // tensor_dim0_stride = K/2 units (= K*2 bytes)
      0, 0
  };
  i32x4 z4 = {0, 0, 0, 0};
#if defined(__clang_major__) && (__clang_major__ >= 23)
  i32x8 z8 = {0, 0, 0, 0, 0, 0, 0, 0};
  __builtin_amdgcn_tensor_load_to_lds(g0, g1, z4, z4, z8, 0);
#else
  __builtin_amdgcn_tensor_load_to_lds(g0, g1, z4, z4, 0);
#endif
}
#endif

// ---------------------------------------------------------------------------
// Tiled WMMA GEMM: C[M,N] = act(A[M,K] * Bt^T + bias (+pos)) (+res)
//   A:  fp32 row-major [M][K]      (converted to f16 while staging to LDS)
//   Bt: f16  row-major [N][K]      (pre-converted weights, K contiguous)
// 256 threads = 8 waves; wave grid 4(M) x 2(N); per-wave 32x64 -> 2x4 frags.
// Double-buffered LDS; B tiles arrive via TDM when available.
// ---------------------------------------------------------------------------
template <int ACT, bool HAS_BIAS, bool HAS_RES, bool HAS_POS>
__global__ __launch_bounds__(256) void wmma_gemm_k(
    const float* __restrict__ A, const _Float16* __restrict__ Bt,
    float* __restrict__ C, const float* __restrict__ bias,
    const float* __restrict__ res, const float* __restrict__ pos,
    int M, int N, int K,
    long long sA, long long sB, long long sC, long long sBias) {
  __shared__ alignas(16) _Float16 As[2][GEMM_BM][GEMM_LDT];
  __shared__ alignas(16) _Float16 Bs[2][GEMM_BN][GEMM_LDT];

  const int tid  = threadIdx.x;
  const int lane = tid & 31;
  const int wave = tid >> 5;
  const int wm   = (wave >> 1) * 32;
  const int wn   = (wave & 1) * 64;
  const int m0   = blockIdx.y * GEMM_BM;
  const int n0   = blockIdx.x * GEMM_BN;
  const long long z = blockIdx.z;
  A += z * sA;  Bt += z * sB;  C += z * sC;
  if (HAS_BIAS) bias += z * sBias;
  if (HAS_RES)  res  += z * sC;
  (void)M;

  const int arow = tid >> 1, acol = (tid & 1) * 16;   // A: 2 thr/row, 16 floats
  float4 ra[4];
  auto gloadA = [&](int k0) {
    const float4* pa = reinterpret_cast<const float4*>(
        A + (size_t)(m0 + arow) * K + k0 + acol);
#pragma unroll
    for (int i = 0; i < 4; ++i) ra[i] = pa[i];
  };
  auto stageA = [&](int buf) {
    _Float16* pa = &As[buf][arow][acol];
#pragma unroll
    for (int i = 0; i < 4; ++i) {
      h4 h = {(_Float16)ra[i].x, (_Float16)ra[i].y,
              (_Float16)ra[i].z, (_Float16)ra[i].w};
      *reinterpret_cast<h4*>(pa + 4 * i) = h;
    }
  };

#if !USE_TDM
  const int bn = tid & 127, bk = (tid >> 7) * 16;     // B: thr per row, 16 k
  v8h rbh[2];
  auto gloadB = [&](int k0) {
    const v8h* pb = reinterpret_cast<const v8h*>(
        Bt + (size_t)(n0 + bn) * K + k0 + bk);
    rbh[0] = pb[0]; rbh[1] = pb[1];
  };
  auto stageB = [&](int buf) {
    v8h* q = reinterpret_cast<v8h*>(&Bs[buf][bn][bk]);
    q[0] = rbh[0]; q[1] = rbh[1];
  };
#endif

  v8f acc[2][4] = {};
  const int kb   = (lane >> 4) << 3;   // 0/8: 16-bit fragment K layout
  const int rsel = lane & 15;
  auto compute = [&](int buf) {
    HF a[2], b[4];
#pragma unroll
    for (int mt = 0; mt < 2; ++mt) {
      const _Float16* p = &As[buf][wm + mt * 16 + rsel][kb];
      a[mt].h[0] = *reinterpret_cast<const v8h*>(p);
      a[mt].h[1] = *reinterpret_cast<const v8h*>(p + 16);
    }
#pragma unroll
    for (int nt = 0; nt < 4; ++nt) {
      const _Float16* p = &Bs[buf][wn + nt * 16 + rsel][kb];
      b[nt].h[0] = *reinterpret_cast<const v8h*>(p);
      b[nt].h[1] = *reinterpret_cast<const v8h*>(p + 16);
    }
#pragma unroll
    for (int mt = 0; mt < 2; ++mt)
#pragma unroll
      for (int nt = 0; nt < 4; ++nt)
        acc[mt][nt] = __builtin_amdgcn_wmma_f32_16x16x32_f16(
            false, a[mt].v, false, b[nt].v, (short)0, acc[mt][nt], false, false);
  };

  const int KT = K / GEMM_BK;
  gloadA(0);
#if USE_TDM
  if (wave == 0)
    tdm_load_b_tile(Bt + (size_t)n0 * K, (unsigned)(size_t)&Bs[0][0][0], K);
#else
  gloadB(0);
#endif
  stageA(0);
#if USE_TDM
  if (wave == 0) __builtin_amdgcn_s_wait_tensorcnt(0);
#else
  stageB(0);
#endif
  __syncthreads();

  int buf = 0;
  for (int kt = 0; kt < KT; ++kt) {
    const bool has_next = (kt + 1 < KT);
    if (has_next) {
      gloadA((kt + 1) * GEMM_BK);
#if USE_TDM
      if (wave == 0)
        tdm_load_b_tile(Bt + (size_t)n0 * K + (kt + 1) * GEMM_BK,
                        (unsigned)(size_t)&Bs[buf ^ 1][0][0], K);
#else
      gloadB((kt + 1) * GEMM_BK);
#endif
      if (kt + 2 < KT)
        __builtin_prefetch(A + (size_t)(m0 + arow) * K + (kt + 2) * GEMM_BK + acol, 0, 1);
    }
    compute(buf);
    if (has_next) {
      stageA(buf ^ 1);
#if USE_TDM
      if (wave == 0) __builtin_amdgcn_s_wait_tensorcnt(0);
#else
      stageB(buf ^ 1);
#endif
      __syncthreads();
      buf ^= 1;
    }
  }

  // Epilogue: C/D layout -> lanes 0-15: M=j, lanes 16-31: M=j+8; N = lane&15
  const int csel = lane & 15;
  const int roff = (lane >> 4) << 3;
#pragma unroll
  for (int mt = 0; mt < 2; ++mt) {
#pragma unroll
    for (int nt = 0; nt < 4; ++nt) {
#pragma unroll
      for (int j = 0; j < 8; ++j) {
        const int r = m0 + wm + mt * 16 + roff + j;
        const int c = n0 + wn + nt * 16 + csel;
        float v = acc[mt][nt][j];
        if (HAS_BIAS) v += bias[c];
        if (HAS_POS)  v += pos[(size_t)(r & 1023) * N + c];
        v = act_fn<ACT>(v);
        const size_t oi = (size_t)r * N + c;
        if (HAS_RES) v += res[oi];
        C[oi] = v;
      }
    }
  }
}

// ---------------------------------------------------------------------------
// LayerNorm over E=1024, one wave per token (8 tokens / 256-thread block).
// ---------------------------------------------------------------------------
__global__ __launch_bounds__(256) void layernorm_k(
    const float* __restrict__ x, float* __restrict__ y,
    const float* __restrict__ g, const float* __restrict__ b) {
  const int wave = threadIdx.x >> 5, lane = threadIdx.x & 31;
  const size_t tok = (size_t)blockIdx.x * 8 + wave;
  const float* xp = x + tok * 1024;
  float v[32];
  float s = 0.f;
#pragma unroll
  for (int i = 0; i < 32; ++i) { v[i] = xp[lane + i * 32]; s += v[i]; }
#pragma unroll
  for (int o = 16; o; o >>= 1) s += __shfl_xor(s, o, 32);
  const float mean = s * (1.f / 1024.f);
  float q = 0.f;
#pragma unroll
  for (int i = 0; i < 32; ++i) { const float d = v[i] - mean; q += d * d; }
#pragma unroll
  for (int o = 16; o; o >>= 1) q += __shfl_xor(q, o, 32);
  const float rstd = rsqrtf(q * (1.f / 1024.f) + 1e-5f);
  float* yp = y + tok * 1024;
#pragma unroll
  for (int i = 0; i < 32; ++i) {
    const int e = lane + i * 32;
    yp[e] = (v[i] - mean) * rstd * g[e] + b[e];
  }
}

// ---------------------------------------------------------------------------
// Fully-unrolled 32-point radix-2 FFT (constant indices -> registers).
// ---------------------------------------------------------------------------
__device__ __forceinline__ void fft32(float* re, float* im, float sgn) {
#pragma unroll
  for (int i = 0; i < 32; ++i) {
    const int j = ((i & 1) << 4) | ((i & 2) << 2) | (i & 4) | ((i & 8) >> 2) | ((i & 16) >> 4);
    if (j > i) {
      float t = re[i]; re[i] = re[j]; re[j] = t;
      t = im[i]; im[i] = im[j]; im[j] = t;
    }
  }
#pragma unroll
  for (int s = 1; s <= 5; ++s) {
    const int len = 1 << s, half = len >> 1;
#pragma unroll
    for (int j = 0; j < half; ++j) {
      float c, sn;
      __sincosf(sgn * 6.28318530718f * (float)j / (float)len, &sn, &c);
#pragma unroll
      for (int i0 = j; i0 < 32; i0 += len) {
        const int a = i0, b = i0 + half;
        const float tr = re[b] * c - im[b] * sn;
        const float ti = re[b] * sn + im[b] * c;
        re[b] = re[a] - tr; im[b] = im[a] - ti;
        re[a] += tr;        im[a] += ti;
      }
    }
  }
}

// rfft2(32x32, ortho) per (b,e) plane; 4 waves/block, wave per plane.
// Packed for the doubled-real frequency GEMM:
// freqA[(n*4352 + pos)*128 + i] = Re, [...+64+i] = Im,  pos = b*544 + h*17 + w.
__global__ __launch_bounds__(128) void fft_fwd_k(
    const float* __restrict__ tok, float* __restrict__ freqA) {
  __shared__ float lre[4][32][33];
  __shared__ float lim[4][32][33];
  const int wave = threadIdx.x >> 5, lane = threadIdx.x & 31;
  const int plane = blockIdx.x * 4 + wave;     // 0..8191
  const int bb = plane >> 10, e = plane & 1023;
  float re[32], im[32];
  {  // W-direction FFT, lane = row h
    const float* p = tok + ((size_t)(bb * 1024 + lane * 32) * 1024 + e);
#pragma unroll
    for (int w = 0; w < 32; ++w) { re[w] = p[(size_t)w * 1024]; im[w] = 0.f; }
    fft32(re, im, -1.f);
#pragma unroll
    for (int w = 0; w < 32; ++w) { lre[wave][lane][w] = re[w]; lim[wave][lane][w] = im[w]; }
  }
  __syncthreads();
  if (lane < 17) {  // H-direction FFT for kept mode w = lane
#pragma unroll
    for (int h = 0; h < 32; ++h) { re[h] = lre[wave][h][lane]; im[h] = lim[wave][h][lane]; }
    fft32(re, im, -1.f);
    const int n = e >> 6, i = e & 63;
#pragma unroll
    for (int h = 0; h < 32; ++h) {
      const size_t pos = (size_t)bb * 544 + h * 17 + lane;
      float* d = freqA + ((size_t)n * 4352 + pos) * 128;
      d[i]      = re[h] * (1.f / 32.f);   // ortho: 1/sqrt(1024)
      d[64 + i] = im[h] * (1.f / 32.f);
    }
  }
}

// irfft2 (ortho) + bias(lnx) + residual(tok0), in place into tok0.
__global__ __launch_bounds__(128) void fft_inv_k(
    const float* __restrict__ freqY, const float* __restrict__ lnx,
    float* __restrict__ tok0) {
  __shared__ float lre[4][32][33];
  __shared__ float lim[4][32][33];
  const int wave = threadIdx.x >> 5, lane = threadIdx.x & 31;
  const int plane = blockIdx.x * 4 + wave;
  const int bb = plane >> 10, e = plane & 1023;
  const int n = e >> 6, i = e & 63;
  float re[32], im[32];
  if (lane < 17) {  // inverse along H for w = lane
#pragma unroll
    for (int h = 0; h < 32; ++h) {
      const float* s = freqY + ((size_t)n * 4352 + (size_t)bb * 544 + h * 17 + lane) * 128;
      re[h] = s[i]; im[h] = s[64 + i];
    }
    fft32(re, im, 1.f);
#pragma unroll
    for (int h = 0; h < 32; ++h) { lre[wave][h][lane] = re[h]; lim[wave][h][lane] = im[h]; }
  }
  __syncthreads();
  {  // inverse along W (Hermitian extension of 17 modes), lane = row h
#pragma unroll
    for (int w = 0; w < 17; ++w) { re[w] = lre[wave][lane][w]; im[w] = lim[wave][lane][w]; }
#pragma unroll
    for (int w = 17; w < 32; ++w) { re[w] = re[32 - w]; im[w] = -im[32 - w]; }
    fft32(re, im, 1.f);
    const size_t base = (size_t)(bb * 1024 + lane * 32) * 1024 + e;
#pragma unroll
    for (int w = 0; w < 32; ++w) {
      const size_t idx = base + (size_t)w * 1024;
      tok0[idx] = tok0[idx] + lnx[idx] + re[w] * (1.f / 32.f);  // ortho inverse
    }
  }
}

// ---------------------------------------------------------------------------
// Prep / data-movement kernels
// ---------------------------------------------------------------------------
__global__ __launch_bounds__(256) void im2col_k(
    const float* __restrict__ x, float* __restrict__ Aim) {
  const int row = blockIdx.x;                        // b*1024 + gh*32 + gw
  const int bb = row >> 10, t = row & 1023, gh = t >> 5, gw = t & 31;
  for (int k = threadIdx.x; k < 512; k += 256) {
    const int c = k >> 8, p1 = (k >> 4) & 15, p2 = k & 15;
    Aim[(size_t)row * 512 + k] =
        x[((size_t)(bb * 2 + c) * 512 + gh * 16 + p1) * 512 + gw * 16 + p2];
  }
}

// f32 -> f16 copy (already [N][K] layout, e.g. patch_w (E, C*P*P)).
__global__ __launch_bounds__(256) void cvt_copy_h_k(
    const float* __restrict__ s, _Float16* __restrict__ d) {
  const size_t idx = (size_t)blockIdx.x * 256 + threadIdx.x;
  d[idx] = (_Float16)s[idx];
}

// f32 [K][N] -> f16 [N][K] transpose+convert; K = 1<<kshift.
__global__ __launch_bounds__(256) void cvt_trans_h_k(
    const float* __restrict__ s, _Float16* __restrict__ d, int kshift, int N) {
  const size_t idx = (size_t)blockIdx.x * 256 + threadIdx.x;   // over N*K
  const int K = 1 << kshift;
  const int n = (int)(idx >> kshift), k = (int)(idx & (K - 1));
  d[idx] = (_Float16)s[(size_t)k * N + n];
}

// Doubled-real complex weights -> f16 [n][c][r] ([[Wr,Wi],[-Wi,Wr]] transposed
// to N-major for the GEMM's Bt layout), plus doubled f32 biases [br|bi].
__global__ __launch_bounds__(256) void prep_freqw_k(
    const float* __restrict__ w1, const float* __restrict__ b1,
    const float* __restrict__ w2, const float* __restrict__ b2,
    _Float16* __restrict__ W1dh, _Float16* __restrict__ W2dh,
    float* __restrict__ b1d, float* __restrict__ b2d) {
  const int idx = blockIdx.x * 256 + threadIdx.x;  // 262144 = 16*128*128
  const int n = idx >> 14, c = (idx >> 7) & 127, r = idx & 127;
  const int i = r & 63, o = c & 63, ir = r >> 6, ic = c >> 6;
  int ch; float sgn;
  if (!ir && !ic)      { ch = 0; sgn = 1.f;  }
  else if (!ir && ic)  { ch = 1; sgn = 1.f;  }
  else if (ir && !ic)  { ch = 1; sgn = -1.f; }
  else                 { ch = 0; sgn = 1.f;  }
  const size_t wi = ((size_t)(ch * 16 + n) * 64 + i) * 64 + o;
  W1dh[idx] = (_Float16)(sgn * w1[wi]);
  W2dh[idx] = (_Float16)(sgn * w2[wi]);
  if (idx < 2048) {
    const int n2 = idx >> 7, c2 = idx & 127, cc = c2 >> 6;
    b1d[idx] = b1[(size_t)(cc * 16 + n2) * 64 + (c2 & 63)];
    b2d[idx] = b2[(size_t)(cc * 16 + n2) * 64 + (c2 & 63)];
  }
}

// 'b h w (p1 p2 c) -> b c (h p1) (w p2)'
__global__ __launch_bounds__(256) void head_rearrange_k(
    const float* __restrict__ Hout, float* __restrict__ out) {
  const size_t idx = (size_t)blockIdx.x * 256 + threadIdx.x;  // 4194304
  const int xq = (int)(idx & 511);
  size_t t = idx >> 9;
  const int y = (int)(t & 511); t >>= 9;
  const int c = (int)(t & 1);
  const int bb = (int)(t >> 1);
  const int gh = y >> 4, p1 = y & 15, gw = xq >> 4, p2 = xq & 15;
  out[idx] = Hout[(size_t)(bb * 1024 + gh * 32 + gw) * 512 + (p1 * 32 + p2 * 2 + c)];
}

// ---------------------------------------------------------------------------
// GEMM dispatch
// ---------------------------------------------------------------------------
enum { M_PATCH, M_RELU, M_SSHR, M_GELU, M_RES, M_SIG };

static void launch_gemm(hipStream_t s, const float* A, const _Float16* Bt,
                        float* C, const float* bias, const float* res,
                        const float* pos, int M, int N, int K, int mode,
                        int batches, long long sA, long long sB, long long sC,
                        long long sBias) {
  dim3 grid(N / GEMM_BN, M / GEMM_BM, batches), blk(256);
  switch (mode) {
    case M_PATCH:
      wmma_gemm_k<ACT_NONE, true, false, true><<<grid, blk, 0, s>>>(
          A, Bt, C, bias, res, pos, M, N, K, sA, sB, sC, sBias); break;
    case M_RELU:
      wmma_gemm_k<ACT_RELU, true, false, false><<<grid, blk, 0, s>>>(
          A, Bt, C, bias, res, pos, M, N, K, sA, sB, sC, sBias); break;
    case M_SSHR:
      wmma_gemm_k<ACT_SSHR, true, false, false><<<grid, blk, 0, s>>>(
          A, Bt, C, bias, res, pos, M, N, K, sA, sB, sC, sBias); break;
    case M_GELU:
      wmma_gemm_k<ACT_GELU, true, false, false><<<grid, blk, 0, s>>>(
          A, Bt, C, bias, res, pos, M, N, K, sA, sB, sC, sBias); break;
    case M_RES:
      wmma_gemm_k<ACT_NONE, true, true, false><<<grid, blk, 0, s>>>(
          A, Bt, C, bias, res, pos, M, N, K, sA, sB, sC, sBias); break;
    default:
      wmma_gemm_k<ACT_SIG, false, false, false><<<grid, blk, 0, s>>>(
          A, Bt, C, bias, res, pos, M, N, K, sA, sB, sC, sBias); break;
  }
}

// ---------------------------------------------------------------------------
// kernel_launch: full forward pass.
// Workspace (floats): tok0(8M) | tok1(8M) | big(32M, time-shared: im2col /
// freqA+freqH+freqY / MLP hidden / head out) | b1d | b2d | f16 weight buffers.
// Total ~ 216 MB.
// ---------------------------------------------------------------------------
extern "C" void kernel_launch(void* const* d_in, const int* in_sizes, int n_in,
                              void* d_out, int out_size, void* d_ws, size_t ws_size,
                              hipStream_t stream) {
  (void)in_sizes; (void)out_size; (void)ws_size;
  const float* x       = (const float*)d_in[0];
  const float* patch_w = (const float*)d_in[1];
  const float* patch_b = (const float*)d_in[2];
  const float* pos     = (const float*)d_in[3];
  const float* pred_w  = (const float*)d_in[n_in - 1];

  float* ws = (float*)d_ws;
  const size_t TOK  = 8192ull * 1024;        // 8,388,608
  const size_t BIG  = 8192ull * 4096;        // 33,554,432
  const size_t FREQ = 16ull * 4352 * 128;    // 8,912,896
  float* tok0 = ws;
  float* tok1 = tok0 + TOK;
  float* big  = tok1 + TOK;
  float* b1d  = big + BIG;
  float* b2d  = b1d + 2048;
  _Float16* BtPatch = (_Float16*)(b2d + 2048);     // [1024][512]
  _Float16* BtPred  = BtPatch + 524288;            // [512][1024]
  _Float16* BtM1    = BtPred + 524288;             // [4096][1024]
  _Float16* BtM2    = BtM1 + 4194304;              // [1024][4096]
  _Float16* W1dh    = BtM2 + 4194304;              // 16 x [128][128]
  _Float16* W2dh    = W1dh + 262144;
  float* freqA = big;
  float* freqH = big + FREQ;
  float* freqY = big + 2 * FREQ;

  // --- one-time weight conversion + patch embed ---
  cvt_copy_h_k<<<2048, 256, 0, stream>>>(patch_w, BtPatch);        // already [N][K]
  cvt_trans_h_k<<<2048, 256, 0, stream>>>(pred_w, BtPred, 10, 512);
  im2col_k<<<8192, 256, 0, stream>>>(x, big);
  launch_gemm(stream, big, BtPatch, tok0, patch_b, nullptr, pos,
              8192, 1024, 512, M_PATCH, 1, 0, 0, 0, 0);

  // --- 6 AFNO blocks ---
  for (int L = 0; L < 6; ++L) {
    const int base = 4 + L * 12;
    const float* ln1g = (const float*)d_in[base + 0];
    const float* ln1b = (const float*)d_in[base + 1];
    const float* w1   = (const float*)d_in[base + 2];
    const float* b1   = (const float*)d_in[base + 3];
    const float* w2   = (const float*)d_in[base + 4];
    const float* b2   = (const float*)d_in[base + 5];
    const float* ln2g = (const float*)d_in[base + 6];
    const float* ln2b = (const float*)d_in[base + 7];
    const float* mw1  = (const float*)d_in[base + 8];
    const float* mb1  = (const float*)d_in[base + 9];
    const float* mw2  = (const float*)d_in[base + 10];
    const float* mb2  = (const float*)d_in[base + 11];

    // per-layer weight prep (f16, N-major)
    prep_freqw_k<<<1024, 256, 0, stream>>>(w1, b1, w2, b2, W1dh, W2dh, b1d, b2d);
    cvt_trans_h_k<<<16384, 256, 0, stream>>>(mw1, BtM1, 10, 4096);
    cvt_trans_h_k<<<16384, 256, 0, stream>>>(mw2, BtM2, 12, 1024);

    layernorm_k<<<1024, 256, 0, stream>>>(tok0, tok1, ln1g, ln1b);
    fft_fwd_k<<<2048, 128, 0, stream>>>(tok1, freqA);
    // frequency block-MLP as 16 batched real GEMMs (complex folded)
    launch_gemm(stream, freqA, W1dh, freqH, b1d, nullptr, nullptr,
                4352, 128, 128, M_RELU, 16, 4352 * 128, 16384, 4352 * 128, 128);
    launch_gemm(stream, freqH, W2dh, freqY, b2d, nullptr, nullptr,
                4352, 128, 128, M_SSHR, 16, 4352 * 128, 16384, 4352 * 128, 128);
    fft_inv_k<<<2048, 128, 0, stream>>>(freqY, tok1, tok0);  // + bias + residual

    layernorm_k<<<1024, 256, 0, stream>>>(tok0, tok1, ln2g, ln2b);
    launch_gemm(stream, tok1, BtM1, big, mb1, nullptr, nullptr,
                8192, 4096, 1024, M_GELU, 1, 0, 0, 0, 0);
    launch_gemm(stream, big, BtM2, tok0, mb2, tok0, nullptr,
                8192, 1024, 4096, M_RES, 1, 0, 0, 0, 0);
  }

  // --- head: sigmoid(x @ pred_w), then pixel-shuffle rearrange ---
  launch_gemm(stream, tok0, BtPred, big, nullptr, nullptr, nullptr,
              8192, 512, 1024, M_SIG, 1, 0, 0, 0, 0);
  head_rearrange_k<<<16384, 256, 0, stream>>>(big, (float*)d_out);
}
